// Qwen3_5Attention_71433896067218
// MI455X (gfx1250) — matrix operand order repaired and checked by reading, not compiled, not run on hardware
//
#include <hip/hip_runtime.h>
#include <hip/hip_bf16.h>

// ---------------------------------------------------------------------------
// Problem constants (from the reference)
// ---------------------------------------------------------------------------
#define B_ 2
#define S_ 2048
#define M_ 2048
#define U_ 16
#define V_ 4
#define D_ 128
#define ROT_ 32

typedef __bf16 bf16_t;
typedef __bf16 v8bf  __attribute__((ext_vector_type(8)));
typedef __bf16 v16bf __attribute__((ext_vector_type(16)));
typedef float  v8f   __attribute__((ext_vector_type(8)));

union Frag16 { v16bf v; v8bf h[2]; };

// ---------------- CDNA5 feature guards (compile-safe fallbacks) ------------
#if defined(__HIP_DEVICE_COMPILE__) && __has_builtin(__builtin_amdgcn_global_load_async_to_lds_b128) && __has_builtin(__builtin_amdgcn_s_wait_asynccnt)
#define HAVE_ASYNC_LDS 1
#else
#define HAVE_ASYNC_LDS 0
#endif

#if defined(__HIP_DEVICE_COMPILE__) && __has_builtin(__builtin_amdgcn_tensor_load_to_lds) && __has_builtin(__builtin_amdgcn_s_wait_tensorcnt)
#define HAVE_TDM 1
#else
#define HAVE_TDM 0
#endif

#if __has_include(<hip/amd_detail/amd_gfx1250_TDM.h>)
#define TDM_6ARG 1      // amdgpu-toolchain (clang-23 / therock-10.0 headers)
#else
#define TDM_6ARG 0      // ROCm 7.2 (clang-22): 5-arg builtin
#endif

typedef unsigned int u32x4 __attribute__((ext_vector_type(4)));
typedef int          i32x4 __attribute__((ext_vector_type(4)));
typedef int          i32x8 __attribute__((ext_vector_type(8)));

#define AS_GLOBAL __attribute__((address_space(1)))
#define AS_LDS    __attribute__((address_space(3)))

__device__ __forceinline__ bf16_t f2bf(float f) {
    union { float f; unsigned u; } x; x.f = f;
    unsigned r = x.u + 0x7FFFu + ((x.u >> 16) & 1u);   // round-to-nearest-even
    unsigned short hs = (unsigned short)(r >> 16);
    return __builtin_bit_cast(bf16_t, hs);
}

#if HAVE_TDM
// Issue a 2D TDM tile load: global (row-major, stride0 elems of 2B) -> LDS.
// D# bitfields per CDNA5 ISA ch.8 (group0: count/lds/global addr/type,
// group1: data_size=2B, tensor dims, tile dims, dim0 stride).
__device__ __forceinline__ void tdm_load_2d_bf16(unsigned lds_off,
                                                 unsigned long long gaddr,
                                                 unsigned tensor_d0, unsigned tensor_d1,
                                                 unsigned tile_d0,   unsigned tile_d1,
                                                 unsigned stride0) {
    u32x4 g0;
    g0[0] = 1u;                                            // count=1, user mode
    g0[1] = lds_off;                                       // lds_addr (bytes)
    g0[2] = (unsigned)(gaddr & 0xffffffffu);               // global_addr[31:0]
    g0[3] = (unsigned)((gaddr >> 32) & 0x1ffffffu)         // global_addr[56:32]
          | (2u << 30);                                    // type = 2 ("image")
    i32x8 g1;
    g1[0] = (int)(1u << 16);                               // data_size = 1 -> 2 bytes
    g1[1] = (int)((tensor_d0 & 0xffffu) << 16);            // tensor_dim0[15:0]
    g1[2] = (int)((tensor_d0 >> 16) | ((tensor_d1 & 0xffffu) << 16));
    g1[3] = (int)((tensor_d1 >> 16) | (tile_d0 << 16));    // tile_dim0
    g1[4] = (int)tile_d1;                                  // tile_dim1 (tile_dim2=0)
    g1[5] = (int)stride0;                                  // tensor_dim0_stride lo32
    g1[6] = 0;
    g1[7] = 0;
    i32x4 z4 = {0, 0, 0, 0};
#if TDM_6ARG
    i32x8 z8 = {0, 0, 0, 0, 0, 0, 0, 0};
    __builtin_amdgcn_tensor_load_to_lds(g0, g1, z4, z4, z8, 0);
#else
    __builtin_amdgcn_tensor_load_to_lds(g0, g1, z4, z4, 0);
#endif
}
#endif // HAVE_TDM

// ---------------------------------------------------------------------------
// K0: fp32 -> bf16 cast (n multiple of 4)
// ---------------------------------------------------------------------------
__global__ __launch_bounds__(256) void cast_f32_bf16(const float* __restrict__ src,
                                                     bf16_t* __restrict__ dst, int n) {
    int i = (blockIdx.x * 256 + threadIdx.x) * 4;
    if (i + 3 < n) {
        float4 f = *(const float4*)(src + i);
        dst[i + 0] = f2bf(f.x);
        dst[i + 1] = f2bf(f.y);
        dst[i + 2] = f2bf(f.z);
        dst[i + 3] = f2bf(f.w);
    }
}

// ---------------------------------------------------------------------------
// K1: tiled bf16 WMMA GEMM   C[MrowsxN](f32) = A[MrowsxK](bf16) * B[KxN](bf16)
//     tile 128x128, K-step 64, 8 waves (2x4), 16 v_wmma per K-step per wave
// ---------------------------------------------------------------------------
#define BM 128
#define BN 128
#define BK 64

__global__ __launch_bounds__(256) void gemm_bf16(const bf16_t* __restrict__ A,
                                                 const bf16_t* __restrict__ B,
                                                 float* __restrict__ C,
                                                 int Mrows, int N, int K) {
    __shared__ __align__(32) bf16_t As[BM][BK];     // m-major
    __shared__ __align__(32) bf16_t Bst[BN][BK];    // n-major (transposed B tile)

    const int tid  = threadIdx.x;
    const int lane = tid & 31;
    const int wid  = tid >> 5;
    const int wm   = wid & 1;     // 2 wave rows
    const int wn   = wid >> 1;    // 4 wave cols
    const int m0   = blockIdx.y * BM;
    const int n0   = blockIdx.x * BN;

    v8f acc[4][2];
#pragma unroll
    for (int mi = 0; mi < 4; ++mi)
#pragma unroll
        for (int ni = 0; ni < 2; ++ni) acc[mi][ni] = (v8f)0.0f;

    const int ar = tid >> 1, ac = (tid & 1) * 32;       // A staging map
    const int bk_r = tid & 63, bn_c = (tid >> 6) * 32;  // B staging map

    for (int bk = 0; bk < K; bk += BK) {
        // --- stage A tile (128x64): 64B per thread, contiguous
        {
            const bf16_t* g = A + (size_t)(m0 + ar) * K + bk + ac;
#if HAVE_ASYNC_LDS
            bf16_t* gnc = const_cast<bf16_t*>(g);
#pragma unroll
            for (int j = 0; j < 4; ++j)
                __builtin_amdgcn_global_load_async_to_lds_b128(
                    (AS_GLOBAL i32x4*)(gnc + j * 8),
                    (AS_LDS i32x4*)&As[ar][ac + j * 8], 0, 0);
#else
#pragma unroll
            for (int j = 0; j < 4; ++j)
                *(v8bf*)&As[ar][ac + j * 8] = *(const v8bf*)(g + j * 8);
#endif
        }
        // --- stage B tile (64x128) transposed into Bst[n][k]
        {
            const bf16_t* g = B + (size_t)(bk + bk_r) * N + n0 + bn_c;
#pragma unroll
            for (int q = 0; q < 4; ++q) {
                v8bf bv = *(const v8bf*)(g + q * 8);
#pragma unroll
                for (int j = 0; j < 8; ++j)
                    Bst[bn_c + q * 8 + j][bk_r] = bv[j];
            }
        }
        // prefetch next K-tile (global_prefetch_b8)
        if (bk + BK < K) {
            __builtin_prefetch(A + (size_t)(m0 + ar) * K + bk + BK + ac, 0, 1);
            __builtin_prefetch(B + (size_t)(bk + BK + bk_r) * N + n0 + bn_c, 0, 1);
        }
#if HAVE_ASYNC_LDS
        __builtin_amdgcn_s_wait_asynccnt(0);
#endif
        __syncthreads();

        const int base = (lane < 16) ? 0 : 8;    // A-frag half select
        const int kb   = (lane < 16) ? 0 : 8;    // B-frag half select
#pragma unroll
        for (int kc = 0; kc < 2; ++kc) {
            Frag16 afr[4];
#pragma unroll
            for (int mi = 0; mi < 4; ++mi) {
                int row = wm * 64 + mi * 16 + (lane & 15);
                afr[mi].h[0] = *(const v8bf*)&As[row][kc * 32 + base];
                afr[mi].h[1] = *(const v8bf*)&As[row][kc * 32 + 16 + base];
            }
#pragma unroll
            for (int ni = 0; ni < 2; ++ni) {
                int col = wn * 32 + ni * 16 + (lane & 15);
                Frag16 bfr;
                bfr.h[0] = *(const v8bf*)&Bst[col][kc * 32 + kb];
                bfr.h[1] = *(const v8bf*)&Bst[col][kc * 32 + kb + 16];
#pragma unroll
                for (int mi = 0; mi < 4; ++mi)
                    acc[mi][ni] = __builtin_amdgcn_wmma_f32_16x16x32_bf16(
                        false, afr[mi].v, false, bfr.v, (short)0, acc[mi][ni], false, false);
            }
        }
        __syncthreads();
    }

    // --- epilogue: C layout VGPR i -> M=i (lanes 0-15) / M=i+8 (lanes 16-31)
    const int halfOff = (lane < 16) ? 0 : 8;
#pragma unroll
    for (int mi = 0; mi < 4; ++mi)
#pragma unroll
        for (int ni = 0; ni < 2; ++ni) {
            int row = m0 + wm * 64 + mi * 16 + halfOff;
            int col = n0 + wn * 32 + ni * 16 + (lane & 15);
#pragma unroll
            for (int i = 0; i < 8; ++i)
                C[(size_t)(row + i) * N + col] = acc[mi][ni][i];
        }
}

// ---------------------------------------------------------------------------
// K2a: Q RMSNorm + partial RoPE   qg(B,S,U,2D) f32 -> q (B,U,S,D) bf16
// ---------------------------------------------------------------------------
__global__ __launch_bounds__(128) void q_norm_rope(const float* __restrict__ qg,
                                                   const float* __restrict__ qscale,
                                                   bf16_t* __restrict__ qout) {
    __shared__ float sh[128];
    __shared__ float ssum[4];
    int idx = blockIdx.x;
    int u = idx & 15, s = (idx >> 4) & 2047, b = idx >> 15;
    int d = threadIdx.x;

    float x = qg[(((size_t)(b * S_ + s) * U_) + u) * (2 * D_) + d];
    float p = x * x;
#pragma unroll
    for (int off = 16; off >= 1; off >>= 1) p += __shfl_xor(p, off, 32);
    if ((d & 31) == 0) ssum[d >> 5] = p;
    __syncthreads();
    float var = (ssum[0] + ssum[1] + ssum[2] + ssum[3]) * (1.0f / 128.0f);
    float xn  = x * rsqrtf(var + 1e-6f) * qscale[d];
    sh[d] = xn;
    __syncthreads();

    float out = xn;
    if (d < ROT_) {
        float pos = (float)s;
        int j = (d < 16) ? d : d - 16;
        float f = pos * __powf(10000.0f, -(float)j * (1.0f / 16.0f));
        float c = __cosf(f), sn = __sinf(f);
        out = (d < 16) ? (xn * c - sh[d + 16] * sn) : (xn * c + sh[d - 16] * sn);
    }
    qout[(((size_t)(b * U_ + u) * S_) + s) * D_ + d] = f2bf(out);
}

// K2b: K RMSNorm + partial RoPE   k(B,S,V,D) f32 -> k (B,V,S,D) bf16
__global__ __launch_bounds__(128) void k_norm_rope(const float* __restrict__ kraw,
                                                   const float* __restrict__ kscale,
                                                   bf16_t* __restrict__ kout) {
    __shared__ float sh[128];
    __shared__ float ssum[4];
    int idx = blockIdx.x;
    int v = idx & 3, s = (idx >> 2) & 2047, b = idx >> 13;
    int d = threadIdx.x;

    float x = kraw[(((size_t)(b * S_ + s) * V_) + v) * D_ + d];
    float p = x * x;
#pragma unroll
    for (int off = 16; off >= 1; off >>= 1) p += __shfl_xor(p, off, 32);
    if ((d & 31) == 0) ssum[d >> 5] = p;
    __syncthreads();
    float var = (ssum[0] + ssum[1] + ssum[2] + ssum[3]) * (1.0f / 128.0f);
    float xn  = x * rsqrtf(var + 1e-6f) * kscale[d];
    sh[d] = xn;
    __syncthreads();

    float out = xn;
    if (d < ROT_) {
        float pos = (float)s;
        int j = (d < 16) ? d : d - 16;
        float f = pos * __powf(10000.0f, -(float)j * (1.0f / 16.0f));
        float c = __cosf(f), sn = __sinf(f);
        out = (d < 16) ? (xn * c - sh[d + 16] * sn) : (xn * c + sh[d - 16] * sn);
    }
    kout[(((size_t)(b * V_ + v) * S_) + s) * D_ + d] = f2bf(out);
}

// K2c: V repack   v(B,S,V,D) f32 -> (B,V,S,D) bf16
__global__ __launch_bounds__(256) void v_pack(const float* __restrict__ vraw,
                                              bf16_t* __restrict__ vout) {
    int t = blockIdx.x * 256 + threadIdx.x;
    int d = t & 127, v = (t >> 7) & 3, s = (t >> 9) & 2047, b = t >> 20;
    vout[(((size_t)(b * V_ + v) * S_) + s) * D_ + d] = f2bf(vraw[t]);
}

// ---------------------------------------------------------------------------
// K3: flash attention (causal), 4 waves / block, 64 q rows per block
//     K tile staged via TDM tensor_load_to_lds when available
// ---------------------------------------------------------------------------
__global__ __launch_bounds__(128) void flash_attn(const bf16_t* __restrict__ Q,
                                                  const bf16_t* __restrict__ Kb,
                                                  const bf16_t* __restrict__ Vb,
                                                  float* __restrict__ Ob) {
    __shared__ __align__(32) bf16_t Ks[64][128];    // [key][d]
    __shared__ __align__(32) bf16_t Vt[128][64];    // [d][key] (transposed)
    __shared__ __align__(32) bf16_t Ps[4][16][64];  // per-wave P tile

    const int tid = threadIdx.x, lane = tid & 31, w = tid >> 5;
    const int qblk = blockIdx.x;
    const int bh = blockIdx.y;
    const int b = bh >> 4, u = bh & 15, v = u >> 2;    // U/V = 4
    const int q0 = qblk * 64, qw0 = q0 + w * 16;
    const float inv_norm = 0.08838834764831845f;       // 1/sqrt(128)
    const int halfOff = (lane < 16) ? 0 : 8;
    const int base    = (lane < 16) ? 0 : 8;   // A-frag half select
    const int kb16    = (lane < 16) ? 0 : 8;   // B-frag half select

    const size_t kvHead = ((size_t)(b * V_ + v)) * S_;

    // Q fragments: 16 rows x 128 d, 4 chunks of K=32
    Frag16 qf[4];
    {
        int row = qw0 + (lane & 15);
        const bf16_t* qp = Q + (((size_t)(b * U_ + u) * S_) + row) * D_;
#pragma unroll
        for (int c = 0; c < 4; ++c) {
            qf[c].h[0] = *(const v8bf*)(qp + c * 32 + base);
            qf[c].h[1] = *(const v8bf*)(qp + c * 32 + 16 + base);
        }
    }

    v8f o[8];
#pragma unroll
    for (int dt = 0; dt < 8; ++dt) o[dt] = (v8f)0.0f;
    float mrow[8], lrow[8];
#pragma unroll
    for (int i = 0; i < 8; ++i) { mrow[i] = -1e30f; lrow[i] = 0.0f; }

    for (int kt = 0; kt <= qblk; ++kt) {
        __syncthreads();
        // ---- stage K tile (64 x 128 bf16)
#if HAVE_TDM
        if (w == 0) {
            unsigned lds_off = (unsigned)(size_t)(AS_LDS bf16_t*)&Ks[0][0];
            unsigned long long ga =
                (unsigned long long)(size_t)(Kb + (kvHead + kt * 64) * D_);
            // tensor: 128 (d) x 2048 (keys), tile: 128 x 64, stride0 = 128 elems
            tdm_load_2d_bf16(lds_off, ga, 128u, (unsigned)S_, 128u, 64u, 128u);
            __builtin_amdgcn_s_wait_tensorcnt(0);
        }
#else
        {
            int r = tid >> 1, h = (tid & 1) * 64;
            const bf16_t* kp = Kb + (kvHead + kt * 64 + r) * D_ + h;
#pragma unroll
            for (int j = 0; j < 8; ++j)
                *(v8bf*)&Ks[r][h + j * 8] = *(const v8bf*)(kp + j * 8);
        }
#endif
        // ---- stage V tile transposed (all threads)
        {
            int r = tid >> 1, h = (tid & 1) * 64;
            const bf16_t* vp = Vb + (kvHead + kt * 64 + r) * D_ + h;
#pragma unroll
            for (int j = 0; j < 64; ++j) Vt[h + j][r] = vp[j];
        }
        __syncthreads();

        // ---- scores: 4 sub-blocks of 16 keys, reduce over d (4 x K=32)
        v8f sv[4];
#pragma unroll
        for (int sub = 0; sub < 4; ++sub) {
            v8f s = (v8f)0.0f;
#pragma unroll
            for (int c = 0; c < 4; ++c) {
                Frag16 kf;
                const bf16_t* kp = &Ks[sub * 16 + (lane & 15)][c * 32 + kb16];
                kf.h[0] = *(const v8bf*)kp;
                kf.h[1] = *(const v8bf*)(kp + 16);
                s = __builtin_amdgcn_wmma_f32_16x16x32_bf16(
                        false, qf[c].v, false, kf.v, (short)0, s, false, false);
            }
            sv[sub] = s;
        }

        const bool diag = (kt == qblk);
        // ---- online softmax
#pragma unroll
        for (int i = 0; i < 8; ++i) {
            float mx = mrow[i];
#pragma unroll
            for (int sub = 0; sub < 4; ++sub) {
                float val = sv[sub][i] * inv_norm;
                if (diag) {
                    int qrow = qw0 + i + halfOff;
                    int kcol = kt * 64 + sub * 16 + (lane & 15);
                    if (kcol > qrow) val = -1e9f;
                }
                sv[sub][i] = val;
                mx = fmaxf(mx, val);
            }
#pragma unroll
            for (int off = 8; off >= 1; off >>= 1) mx = fmaxf(mx, __shfl_xor(mx, off, 32));
            float rs = 0.0f;
#pragma unroll
            for (int sub = 0; sub < 4; ++sub) {
                float p = __expf(sv[sub][i] - mx);
                rs += p;
                Ps[w][i + halfOff][sub * 16 + (lane & 15)] = f2bf(p);
            }
#pragma unroll
            for (int off = 8; off >= 1; off >>= 1) rs += __shfl_xor(rs, off, 32);
            float alpha = __expf(mrow[i] - mx);
            lrow[i] = lrow[i] * alpha + rs;
            mrow[i] = mx;
#pragma unroll
            for (int dt = 0; dt < 8; ++dt) o[dt][i] *= alpha;
        }

        // ---- P * V : A = P (16x64 -> 2 chunks of K=32), B = V^T slices
        Frag16 pf[2];
#pragma unroll
        for (int f = 0; f < 2; ++f) {
            int row = lane & 15;
            pf[f].h[0] = *(const v8bf*)&Ps[w][row][f * 32 + base];
            pf[f].h[1] = *(const v8bf*)&Ps[w][row][f * 32 + 16 + base];
        }
#pragma unroll
        for (int dt = 0; dt < 8; ++dt) {
#pragma unroll
            for (int f = 0; f < 2; ++f) {
                Frag16 vf;
                const bf16_t* vp2 = &Vt[dt * 16 + (lane & 15)][f * 32 + kb16];
                vf.h[0] = *(const v8bf*)vp2;
                vf.h[1] = *(const v8bf*)(vp2 + 16);
                o[dt] = __builtin_amdgcn_wmma_f32_16x16x32_bf16(
                            false, pf[f].v, false, vf.v, (short)0, o[dt], false, false);
            }
        }
    }

    // ---- write normalized O
#pragma unroll
    for (int dt = 0; dt < 8; ++dt)
#pragma unroll
        for (int i = 0; i < 8; ++i) {
            int q = qw0 + i + halfOff;
            Ob[(((size_t)(b * U_ + u) * S_) + q) * D_ + dt * 16 + (lane & 15)] =
                o[dt][i] / lrow[i];
        }
}

// ---------------------------------------------------------------------------
// K4: sigmoid gate fuse: attn(B,U,S,D) f32 * sigmoid(gate) -> (B*S, U*D) bf16
// ---------------------------------------------------------------------------
__global__ __launch_bounds__(256) void gate_fuse(const float* __restrict__ attn,
                                                 const float* __restrict__ qg,
                                                 bf16_t* __restrict__ gated) {
    int bs = blockIdx.x;               // b*S + s
    int b = bs >> 11, s = bs & 2047;
#pragma unroll
    for (int t = 0; t < 8; ++t) {
        int col = threadIdx.x + t * 256;     // 0..2047 = u*128 + d
        int u = col >> 7, d = col & 127;
        float a = attn[(((size_t)(b * U_ + u) * S_) + s) * D_ + d];
        float g = qg[(((size_t)bs * U_) + u) * (2 * D_) + D_ + d];
        float sig = 1.0f / (1.0f + __expf(-g));
        gated[(size_t)bs * (U_ * D_) + col] = f2bf(a * sig);
    }
}

// ---------------------------------------------------------------------------
// Host launcher
// ---------------------------------------------------------------------------
extern "C" void kernel_launch(void* const* d_in, const int* in_sizes, int n_in,
                              void* d_out, int out_size, void* d_ws, size_t ws_size,
                              hipStream_t stream) {
    const float* hs = (const float*)d_in[0];
    // d_in[1] = attention_mask (causal; implicit in flash kernel)
    const float* wq = (const float*)d_in[2];
    const float* wk = (const float*)d_in[3];
    const float* wv = (const float*)d_in[4];
    const float* wo = (const float*)d_in[5];
    const float* q_scale = (const float*)d_in[6];
    const float* k_scale = (const float*)d_in[7];
    float* out = (float*)d_out;

    const size_t BS  = (size_t)B_ * S_;                 // 4096
    const size_t nHS = BS * M_;                          // 8.4M
    const size_t nWQ = (size_t)M_ * U_ * 2 * D_;         // 8.4M
    const size_t nWK = (size_t)M_ * V_ * D_;             // 1.05M
    const size_t nWO = (size_t)U_ * D_ * M_;             // 4.2M

    size_t off = 0;
    auto carve = [&](size_t bytes) -> void* {
        void* p = (char*)d_ws + off;
        off += (bytes + 255) & ~(size_t)255;
        return p;
    };
    bf16_t* hsb = (bf16_t*)carve(nHS * 2);
    bf16_t* wqb = (bf16_t*)carve(nWQ * 2);
    bf16_t* wkb = (bf16_t*)carve(nWK * 2);
    bf16_t* wvb = (bf16_t*)carve(nWK * 2);
    bf16_t* wob = (bf16_t*)carve(nWO * 2);
    float*  qg  = (float*)carve(BS * U_ * 2 * D_ * 4);   // 67 MB
    float*  kr  = (float*)carve(BS * V_ * D_ * 4);
    float*  vr  = (float*)carve(BS * V_ * D_ * 4);
    bf16_t* qb  = (bf16_t*)carve(BS * U_ * D_ * 2);
    bf16_t* kbb = (bf16_t*)carve(BS * V_ * D_ * 2);
    bf16_t* vbb = (bf16_t*)carve(BS * V_ * D_ * 2);
    float*  at  = (float*)carve(BS * U_ * D_ * 4);
    bf16_t* gt  = (bf16_t*)carve(BS * U_ * D_ * 2);

    // 1) casts
    cast_f32_bf16<<<nHS / 1024, 256, 0, stream>>>(hs, hsb, (int)nHS);
    cast_f32_bf16<<<nWQ / 1024, 256, 0, stream>>>(wq, wqb, (int)nWQ);
    cast_f32_bf16<<<nWK / 1024, 256, 0, stream>>>(wk, wkb, (int)nWK);
    cast_f32_bf16<<<nWK / 1024, 256, 0, stream>>>(wv, wvb, (int)nWK);
    cast_f32_bf16<<<nWO / 1024, 256, 0, stream>>>(wo, wob, (int)nWO);

    // 2) projections (WMMA GEMMs)
    gemm_bf16<<<dim3((U_ * 2 * D_) / BN, BS / BM), 256, 0, stream>>>(
        hsb, wqb, qg, (int)BS, U_ * 2 * D_, M_);
    gemm_bf16<<<dim3((V_ * D_) / BN, BS / BM), 256, 0, stream>>>(
        hsb, wkb, kr, (int)BS, V_ * D_, M_);
    gemm_bf16<<<dim3((V_ * D_) / BN, BS / BM), 256, 0, stream>>>(
        hsb, wvb, vr, (int)BS, V_ * D_, M_);

    // 3) norm + rope + repack
    q_norm_rope<<<B_ * S_ * U_, 128, 0, stream>>>(qg, q_scale, qb);
    k_norm_rope<<<B_ * S_ * V_, 128, 0, stream>>>(kr, k_scale, kbb);
    v_pack<<<(B_ * S_ * V_ * D_) / 256, 256, 0, stream>>>(vr, vbb);

    // 4) causal flash attention (WMMA + TDM K-tile staging)
    flash_attn<<<dim3(S_ / 64, B_ * U_), 128, 0, stream>>>(qb, kbb, vbb, at);

    // 5) gate fuse
    gate_fuse<<<B_ * S_, 256, 0, stream>>>(at, qg, gt);

    // 6) output projection (WMMA GEMM) -> d_out fp32
    gemm_bf16<<<dim3(M_ / BN, BS / BM), 256, 0, stream>>>(
        gt, wob, out, (int)BS, M_, U_ * D_);
}
